// TorchModel_50826642981464
// MI455X (gfx1250) — compile-verified, matrix operand order checked by reading, not run on hardware
//
#include <hip/hip_runtime.h>

typedef __attribute__((ext_vector_type(16))) __bf16 v16bf;
typedef __attribute__((ext_vector_type(8)))  float  v8f;

#define Bdim 64
#define Sdim 512
#define Hdim 768
#define Cdim 9
#define KCH  (Hdim / 32)          /* 24 K-chunks of 32 */
#define MTILES ((Bdim * Sdim) / 16) /* 2048 tiles of 16 rows */

// ---------------------------------------------------------------------------
// Kernel 1: emissions[b,s,c] = hidden[b,s,:] . W[c,:] + bias[c]
// bf16 hi/lo split GEMM on v_wmma_f32_16x16x32_bf16, fp32 accumulate.
// One 16-row M-tile per wave; N padded 9->16 with zero columns in LDS.
// ---------------------------------------------------------------------------
__global__ __launch_bounds__(256) void emissions_wmma_kernel(
    const float* __restrict__ hidden, const float* __restrict__ W,
    const float* __restrict__ bias, float* __restrict__ emis)
{
    __shared__ v16bf sBhi[KCH][32];
    __shared__ v16bf sBlo[KCH][32];

    const int tid = threadIdx.x;

    // Stage W into LDS in WMMA B-fragment layout, split into bf16 hi + lo.
    // Fragment for lane L: column n = L&15 of B (= row n of W), K mapping:
    // element e -> k = c*32 + koff + e + (e>=8 ? 8 : 0), koff = (L>=16)?8:0.
    for (int s = tid; s < KCH * 32; s += 256) {
        const int c    = s >> 5;
        const int lane = s & 31;
        const int n    = lane & 15;
        const int koff = (lane >= 16) ? 8 : 0;
        v16bf bh, bl;
        #pragma unroll
        for (int e = 0; e < 16; ++e) {
            const int k = c * 32 + koff + e + ((e >= 8) ? 8 : 0);
            float x = (n < Cdim) ? W[n * Hdim + k] : 0.0f;
            __bf16 h = (__bf16)x;
            bh[e] = h;
            bl[e] = (__bf16)(x - (float)h);
        }
        sBhi[c][lane] = bh;
        sBlo[c][lane] = bl;
    }
    __syncthreads();

    const int lane = tid & 31;
    const int wave = tid >> 5;
    const int tile = blockIdx.x * 8 + wave;   // grid sized so tile < MTILES always
    const int koff = (lane >= 16) ? 8 : 0;
    const size_t row = (size_t)tile * 16 + (lane & 15);
    const float* hrow = hidden + row * Hdim;

    v8f acc = {0.f, 0.f, 0.f, 0.f, 0.f, 0.f, 0.f, 0.f};

    for (int c = 0; c < KCH; ++c) {
        // A fragment: 16 fp32 from this lane's row, matching A K-layout.
        const float4* p = (const float4*)(hrow + c * 32 + koff);
        float4 f0 = p[0];  // k = koff+0..3
        float4 f1 = p[1];  // k = koff+4..7
        float4 f2 = p[4];  // k = koff+16..19
        float4 f3 = p[5];  // k = koff+20..23
        float a[16] = {f0.x, f0.y, f0.z, f0.w, f1.x, f1.y, f1.z, f1.w,
                       f2.x, f2.y, f2.z, f2.w, f3.x, f3.y, f3.z, f3.w};
        v16bf ah, al;
        #pragma unroll
        for (int e = 0; e < 16; ++e) {
            float x = a[e];
            __bf16 h = (__bf16)x;
            ah[e] = h;
            al[e] = (__bf16)(x - (float)h);
        }
        v16bf bh = sBhi[c][lane];
        v16bf bl = sBlo[c][lane];
        // x*w ~= hi_a*hi_b + hi_a*lo_b + lo_a*hi_b  (fp32 accumulate)
        acc = __builtin_amdgcn_wmma_f32_16x16x32_bf16(false, ah, false, bh,
                                                      (short)0, acc, false, false);
        acc = __builtin_amdgcn_wmma_f32_16x16x32_bf16(false, ah, false, bl,
                                                      (short)0, acc, false, false);
        acc = __builtin_amdgcn_wmma_f32_16x16x32_bf16(false, al, false, bh,
                                                      (short)0, acc, false, false);
    }

    // C/D layout: lane l holds column n=l&15, rows v + 8*(l>=16), v=0..7.
    const int n  = lane & 15;
    const int r0 = tile * 16 + ((lane >= 16) ? 8 : 0);
    if (n < Cdim) {
        const float bb = bias[n];
        #pragma unroll
        for (int v = 0; v < 8; ++v)
            emis[(size_t)(r0 + v) * Cdim + n] = acc[v] + bb;
    }
}

// ---------------------------------------------------------------------------
// Kernel 2: CRF numerator + forward (logsumexp) recursion. One wave per batch.
// Lanes 0..8 own target tag j; full 9-score vector replicated per lane and
// refreshed each step via lane shuffles.
// ---------------------------------------------------------------------------
__global__ __launch_bounds__(32) void crf_forward_kernel(
    const float* __restrict__ emis, const int* __restrict__ tags,
    const int* __restrict__ mask, const float* __restrict__ st,
    const float* __restrict__ tr, const float* __restrict__ et,
    float* __restrict__ out_per_batch)
{
    const int b    = blockIdx.x;
    const int lane = threadIdx.x;

    const int*   tb = tags + b * Sdim;
    const int*   mb = mask + b * Sdim;
    const float* eb = emis + (size_t)b * Sdim * Cdim;

    // ---- numerator: wave-parallel over time steps ----
    float part = 0.0f;
    int   cnt  = 0;
    for (int t = lane; t < Sdim; t += 32) {
        const int mk = mb[t];
        cnt += mk;
        if (t >= 1 && mk) {
            const int tp = tb[t - 1];
            const int tc = tb[t];
            part += tr[tp * Cdim + tc] + eb[t * Cdim + tc];
        }
    }
    #pragma unroll
    for (int o = 16; o >= 1; o >>= 1) {
        part += __shfl_xor(part, o, 32);
        cnt  += __shfl_xor(cnt, o, 32);
    }
    const int t0   = tb[0];
    const int last = cnt - 1;
    float num = part + st[t0] + eb[t0] + et[tb[last]];

    // ---- forward recursion (log-semiring scan) ----
    const int jj = (lane < Cdim) ? lane : 0;
    float trc[Cdim];
    #pragma unroll
    for (int i = 0; i < Cdim; ++i) trc[i] = tr[i * Cdim + jj];

    float sc[Cdim];
    #pragma unroll
    for (int i = 0; i < Cdim; ++i) sc[i] = st[i] + eb[i];

    for (int t = 1; t < Sdim; ++t) {
        float m = -1e30f;
        #pragma unroll
        for (int i = 0; i < Cdim; ++i) m = fmaxf(m, sc[i] + trc[i]);
        float s = 0.0f;
        #pragma unroll
        for (int i = 0; i < Cdim; ++i) s += __expf(sc[i] + trc[i] - m);
        const float nxt = m + __logf(s) + eb[t * Cdim + jj];
        const int   mk  = mb[t];
        float ns[Cdim];
        #pragma unroll
        for (int i = 0; i < Cdim; ++i) ns[i] = __shfl(nxt, i, 32);
        if (mk) {
            #pragma unroll
            for (int i = 0; i < Cdim; ++i) sc[i] = ns[i];
        }
    }

    if (lane == 0) {
        float m = -1e30f;
        #pragma unroll
        for (int j = 0; j < Cdim; ++j) m = fmaxf(m, sc[j] + et[j]);
        float s = 0.0f;
        #pragma unroll
        for (int j = 0; j < Cdim; ++j) s += __expf(sc[j] + et[j] - m);
        const float denom = m + __logf(s);
        out_per_batch[b] = denom - num;   // -llh for this batch
    }
}

// ---------------------------------------------------------------------------
// Kernel 3: mean over 64 per-batch -llh values -> scalar output.
// ---------------------------------------------------------------------------
__global__ __launch_bounds__(32) void mean_kernel(const float* __restrict__ pb,
                                                  float* __restrict__ out)
{
    const int lane = threadIdx.x;
    float v = pb[lane] + pb[lane + 32];
    #pragma unroll
    for (int o = 16; o >= 1; o >>= 1) v += __shfl_xor(v, o, 32);
    if (lane == 0) out[0] = v * (1.0f / (float)Bdim);
}

// ---------------------------------------------------------------------------
extern "C" void kernel_launch(void* const* d_in, const int* in_sizes, int n_in,
                              void* d_out, int out_size, void* d_ws, size_t ws_size,
                              hipStream_t stream) {
    const float* hidden = (const float*)d_in[0];
    const float* W      = (const float*)d_in[1];
    const float* bias   = (const float*)d_in[2];
    const float* st     = (const float*)d_in[3];
    const float* tr     = (const float*)d_in[4];
    const float* et     = (const float*)d_in[5];
    const int*   tags   = (const int*)d_in[6];
    const int*   mask   = (const int*)d_in[7];

    float* emis = (float*)d_ws;                                  // B*S*C fp32
    float* pb   = emis + (size_t)Bdim * Sdim * Cdim;             // 64 fp32

    emissions_wmma_kernel<<<MTILES / 8, 256, 0, stream>>>(hidden, W, bias, emis);
    crf_forward_kernel<<<Bdim, 32, 0, stream>>>(emis, tags, mask, st, tr, et, pb);
    mean_kernel<<<1, 32, 0, stream>>>(pb, (float*)d_out);
}
